// MultiheadAttention_46462956208610
// MI455X (gfx1250) — compile-verified, matrix operand order checked by reading, not run on hardware
//
#include <hip/hip_runtime.h>

// ---------------------------------------------------------------------------
// MHA block for MI455X (gfx1250, wave32, WMMA).
// B=2, S=2048, E=512, H=8, Dh=64.  ~26 GFLOP, ~40MB HBM traffic -> compute
// bound; bf16 WMMA (v_wmma_f32_16x16x32_bf16) with f32 accumulation.
// Pipeline:
//   k_cvt     : one-pass fp32->bf16 of activations + weights
//   k_proj x3 : QKV projections; block shares A-panel via async LDS staging,
//               2x-unrolled ping-pong pipelined K loop, coalesced epilogue
//   k_attn    : flash attention, online softmax, ping-pong K streaming
//   k_out_ln  : out projection fused with LayerNorm
// ---------------------------------------------------------------------------

#define B_  2
#define S_  2048
#define E_  512
#define H_  8
#define DH_ 64
#define M_  (B_ * S_)   // 4096 rows

typedef __attribute__((ext_vector_type(16))) __bf16 v16bf;
typedef __attribute__((ext_vector_type(8)))  __bf16 v8bf;
typedef __attribute__((ext_vector_type(8)))  float  v8f;

#if defined(__gfx1250__) && __has_builtin(__builtin_amdgcn_global_load_async_to_lds_b128)
#define HAVE_ASYNC_LDS 1
#else
#define HAVE_ASYNC_LDS 0
#endif

__device__ __forceinline__ __bf16 f2bf(float x) { return (__bf16)x; }

__device__ __forceinline__ v16bf mk16(v8bf lo, v8bf hi) {
  v16bf r;
#pragma unroll
  for (int i = 0; i < 8; ++i) { r[i] = lo[i]; r[8 + i] = hi[i]; }
  return r;
}

__device__ __forceinline__ v8bf ld8bf(const __bf16* p) {
  return *reinterpret_cast<const v8bf*>(p);
}

// 16-bit A/B WMMA fragment: element j holds K = base + 16*(j>=8) + 8*half + (j&7)
// -> two contiguous 16-byte loads per fragment.
__device__ __forceinline__ v16bf frag_from_bf16(const __bf16* row, int kk, int half) {
  return mk16(ld8bf(row + kk + half * 8), ld8bf(row + kk + 16 + half * 8));
}

#define WMMA_BF16(A, Bm, C) \
  __builtin_amdgcn_wmma_f32_16x16x32_bf16(false, (A), false, (Bm), (short)0, (C), false, false)

// Stage a 16-row x 512-col bf16 panel (16KB) into LDS, block-cooperative,
// via gfx1250 async-to-LDS DMA (ASYNCcnt tracked).
__device__ __forceinline__ void stage_panel(const __bf16* __restrict__ gsrc,
                                            __bf16* lds) {
  int t = threadIdx.x;
#if HAVE_ASYNC_LDS
  typedef int v4i_g __attribute__((vector_size(16)));
  typedef __attribute__((address_space(1))) v4i_g as1_v4i;
  typedef __attribute__((address_space(3))) v4i_g as3_v4i;
#pragma unroll
  for (int i = 0; i < 4; ++i) {
    int off = (t + i * 256) * 8;             // 16B chunk per lane per op
    __builtin_amdgcn_global_load_async_to_lds_b128(
        (as1_v4i*)(gsrc + off), (as3_v4i*)(lds + off), 0, 0);
  }
#if __has_builtin(__builtin_amdgcn_s_wait_asynccnt)
  __builtin_amdgcn_s_wait_asynccnt(0);
#else
  asm volatile("s_wait_asynccnt 0x0" ::: "memory");
#endif
#else
#pragma unroll
  for (int i = 0; i < 4; ++i) {
    int off = (t + i * 256) * 8;
    *reinterpret_cast<v8bf*>(lds + off) = ld8bf(gsrc + off);
  }
#endif
  __syncthreads();
}

// ---------------------------------------------------------------------------
// Kernel 0: fp32 -> bf16 bulk convert (8 elements / thread)
// ---------------------------------------------------------------------------
__global__ void k_cvt(const float* __restrict__ src, __bf16* __restrict__ dst, int n8) {
  int i = blockIdx.x * blockDim.x + threadIdx.x;
  if (i >= n8) return;
  const float4* p = reinterpret_cast<const float4*>(src) + 2 * (size_t)i;
  float4 a = p[0], b = p[1];
  v8bf r;
  r[0] = f2bf(a.x); r[1] = f2bf(a.y); r[2] = f2bf(a.z); r[3] = f2bf(a.w);
  r[4] = f2bf(b.x); r[5] = f2bf(b.y); r[6] = f2bf(b.z); r[7] = f2bf(b.w);
  *(reinterpret_cast<v8bf*>(dst) + i) = r;
}

// ---------------------------------------------------------------------------
// Kernel 1: projection GEMM  Out = A @ W^T + bias  (M=4096, N=512, K=512)
// Block = 16 output rows x 512 cols; wave tg owns n in [tg*64, tg*64+64).
// A-panel staged in LDS once per block; K-loop 2x unrolled with ping-pong
// fragment sets (no register-copy rotation).
// vmode==0 : store bf16 head-major  Out[b][h][s][d]   (for Q, K)
// vmode==1 : store bf16 transposed  Out[b][h][d][s]   (for V -> PV B-frags)
// ---------------------------------------------------------------------------
__global__ void k_proj(const __bf16* __restrict__ A, const __bf16* __restrict__ W,
                       const float* __restrict__ bias, __bf16* __restrict__ Out,
                       int vmode) {
  __shared__ __align__(16) __bf16 apan[16 * E_];   // 16KB; reused as store stage
  int lane = threadIdx.x & 31;
  int wave = threadIdx.x >> 5;
  int tm = blockIdx.x;                       // 0..255  (M/16)
  int tg = wave;                             // 0..7    (N/64) == head index
  int half = lane >> 4;
  int l15  = lane & 15;

  stage_panel(A + (size_t)tm * 16 * E_, apan);
  const __bf16* arow = apan + l15 * E_;

  const __bf16* wrow[4];
#pragma unroll
  for (int t = 0; t < 4; ++t) wrow[t] = W + (size_t)(tg * 64 + t * 16 + l15) * E_;

  v8f acc[4] = {v8f{}, v8f{}, v8f{}, v8f{}};
  v16bf afA = frag_from_bf16(arow, 0, half);
  v16bf wfA[4];
#pragma unroll
  for (int t = 0; t < 4; ++t) wfA[t] = frag_from_bf16(wrow[t], 0, half);

  for (int kk = 0; kk < E_; kk += 64) {
    // load set B (kk+32; always in-bounds since E_ % 64 == 0)
    v16bf afB = frag_from_bf16(arow, kk + 32, half);
    v16bf wfB[4];
#pragma unroll
    for (int t = 0; t < 4; ++t) wfB[t] = frag_from_bf16(wrow[t], kk + 32, half);
    // compute on set A
#pragma unroll
    for (int t = 0; t < 4; ++t) acc[t] = WMMA_BF16(afA, wfA[t], acc[t]);
    // reload set A for kk+64
    if (kk + 64 < E_) {
      afA = frag_from_bf16(arow, kk + 64, half);
#pragma unroll
      for (int t = 0; t < 4; ++t) wfA[t] = frag_from_bf16(wrow[t], kk + 64, half);
    }
    // compute on set B
#pragma unroll
    for (int t = 0; t < 4; ++t) acc[t] = WMMA_BF16(afB, wfB[t], acc[t]);
  }

  int mg0 = tm * 16;                         // block's first row
  int b  = mg0 >> 11;
  int s0 = mg0 & 2047;
  int h  = tg;

  if (vmode) {
    // V transposed: Vt[b][h][d][s]; 8 consecutive s -> one 16B store per tile
#pragma unroll
    for (int t = 0; t < 4; ++t) {
      int d = t * 16 + l15;
      float bv = bias[h * 64 + d];
      v8bf pk;
#pragma unroll
      for (int r = 0; r < 8; ++r) pk[r] = f2bf(acc[t][r] + bv);
      *reinterpret_cast<v8bf*>(Out + (((size_t)b * H_ + h) * DH_ + d) * S_ +
                               s0 + 8 * half) = pk;
    }
  } else {
    // Qh/Kh [b][h][s][d]: stage 16x64 tile in LDS, then coalesced 16B stores
    __syncthreads();                         // A-panel reads finished block-wide
    __bf16* stg = apan + wave * 16 * DH_;    // 2KB per wave
#pragma unroll
    for (int t = 0; t < 4; ++t) {
      int d = t * 16 + l15;
      float bv = bias[h * 64 + d];
#pragma unroll
      for (int r = 0; r < 8; ++r) stg[(r + 8 * half) * DH_ + d] = f2bf(acc[t][r] + bv);
    }
    asm volatile("s_wait_dscnt 0" ::: "memory");   // wave-local staging fence
#pragma unroll
    for (int i = 0; i < 4; ++i) {
      int idx = i * 32 + lane;               // 128 chunks of 8 bf16
      int row = idx >> 3, ch = idx & 7;
      v8bf pk = ld8bf(stg + row * DH_ + ch * 8);
      *reinterpret_cast<v8bf*>(Out + (((size_t)b * H_ + h) * S_ + (s0 + row)) * DH_ +
                               ch * 8) = pk;
    }
  }
}

// ---------------------------------------------------------------------------
// Kernel 2: flash attention per (b,h,q-tile). One wave owns 16 q rows x Dh=64.
// Online softmax; acc VGPR r of lane l corresponds to row m = r + 8*(l>>4).
// c-loop 2x unrolled with ping-pong K fragment sets; next K tile is issued
// before the current step's softmax VALU so its latency is hidden.
// ---------------------------------------------------------------------------
__global__ void k_attn(const __bf16* __restrict__ Qh, const __bf16* __restrict__ Kh,
                       const __bf16* __restrict__ Vt, __bf16* __restrict__ X) {
  __shared__ __align__(16) __bf16 plds[8 * 16 * DH_];  // 2KB/wave: P + out staging
  int lane = threadIdx.x & 31;
  int wave = threadIdx.x >> 5;
  int tile = blockIdx.x * 8 + wave;          // 2048 tiles
  int bh = tile >> 7;                        // 0..15  (b*8+h)
  int qt = tile & 127;                       // q tile 0..127
  int half = lane >> 4;
  int l15  = lane & 15;
  __bf16* pw = plds + wave * 16 * DH_;

  const __bf16* qrow = Qh + ((size_t)bh * S_ + qt * 16 + l15) * DH_;
  v16bf qa0 = frag_from_bf16(qrow, 0, half);
  v16bf qa1 = frag_from_bf16(qrow, 32, half);

  v8f acc[4] = {v8f{}, v8f{}, v8f{}, v8f{}};
  float rmax[8], rsum[8];
#pragma unroll
  for (int r = 0; r < 8; ++r) { rmax[r] = -1e30f; rsum[r] = 0.f; }

  const __bf16* kbase = Kh + (size_t)bh * S_ * DH_;
  const __bf16* vbase = Vt + (size_t)bh * DH_ * S_;

  // scores + online-softmax + P.V for 32 keys starting at c
  auto kv_step = [&](int c, v16bf kb00, v16bf kb01, v16bf kb10, v16bf kb11) {
    v8f z = {};
    v8f s0 = WMMA_BF16(qa0, kb00, z); s0 = WMMA_BF16(qa1, kb01, s0);
    v8f s1 = WMMA_BF16(qa0, kb10, z); s1 = WMMA_BF16(qa1, kb11, s1);

    // V tiles for this key block: issue before the softmax VALU
    v16bf vbv[4];
#pragma unroll
    for (int t = 0; t < 4; ++t) {
      const __bf16* vr = vbase + (size_t)(t * 16 + l15) * S_ + c;
      vbv[t] = mk16(ld8bf(vr + half * 8), ld8bf(vr + 16 + half * 8));
    }

    float p0[8], p1[8], fac[8];
#pragma unroll
    for (int r = 0; r < 8; ++r) {
      float a = s0[r] * 0.125f;              // SCALE = Dh^-0.5, TEMPERATURE = 1
      float b = s1[r] * 0.125f;
      float mx = fmaxf(a, b);
#pragma unroll
      for (int m = 1; m < 16; m <<= 1) mx = fmaxf(mx, __shfl_xor(mx, m, 32));
      mx = fmaxf(mx, rmax[r]);
      p0[r] = __expf(a - mx);
      p1[r] = __expf(b - mx);
      fac[r] = __expf(rmax[r] - mx);
      rmax[r] = mx;
      float ps = p0[r] + p1[r];
#pragma unroll
      for (int m = 1; m < 16; m <<= 1) ps += __shfl_xor(ps, m, 32);
      rsum[r] = rsum[r] * fac[r] + ps;
    }

    // rescale accumulators, stage P (C-layout -> LDS [m][key]) as bf16
#pragma unroll
    for (int r = 0; r < 8; ++r) {
      acc[0][r] *= fac[r]; acc[1][r] *= fac[r];
      acc[2][r] *= fac[r]; acc[3][r] *= fac[r];
      int m = r + 8 * half;
      pw[m * 32 + l15]      = f2bf(p0[r]);
      pw[m * 32 + 16 + l15] = f2bf(p1[r]);
    }
    asm volatile("s_wait_dscnt 0" ::: "memory");   // wave-local LDS RAW fence

    // read P back in A-fragment layout (row = l&15, K = keys 0..31)
    v16bf pa = frag_from_bf16(pw + l15 * 32, 0, half);
#pragma unroll
    for (int t = 0; t < 4; ++t) acc[t] = WMMA_BF16(pa, vbv[t], acc[t]);
  };

  auto load_k = [&](int c, v16bf& k00, v16bf& k01, v16bf& k10, v16bf& k11) {
    const __bf16* k0 = kbase + (size_t)(c + l15) * DH_;
    const __bf16* k1 = k0 + 16 * DH_;
    k00 = frag_from_bf16(k0, 0, half);  k01 = frag_from_bf16(k0, 32, half);
    k10 = frag_from_bf16(k1, 0, half);  k11 = frag_from_bf16(k1, 32, half);
  };

  v16bf kA00, kA01, kA10, kA11, kB00, kB01, kB10, kB11;
  load_k(0, kA00, kA01, kA10, kA11);

  for (int c = 0; c < S_; c += 64) {
    load_k(c + 32, kB00, kB01, kB10, kB11);  // S_ % 64 == 0, always in-bounds
    kv_step(c, kA00, kA01, kA10, kA11);
    if (c + 64 < S_) load_k(c + 64, kA00, kA01, kA10, kA11);
    kv_step(c + 32, kB00, kB01, kB10, kB11);
  }

  // ---- finalize: divide by row sums; stage tile in LDS; coalesced stores
  int b = bh >> 3, h = bh & 7;
#pragma unroll
  for (int t = 0; t < 4; ++t) {
    int d = t * 16 + l15;
#pragma unroll
    for (int r = 0; r < 8; ++r)
      pw[(r + 8 * half) * DH_ + d] = f2bf(acc[t][r] / rsum[r]);
  }
  asm volatile("s_wait_dscnt 0" ::: "memory");
#pragma unroll
  for (int i = 0; i < 4; ++i) {
    int idx = i * 32 + lane;                 // 128 chunks of 8 bf16
    int row = idx >> 3, ch = idx & 7;
    v8bf pk = ld8bf(pw + row * DH_ + ch * 8);
    size_t grow = (size_t)b * S_ + qt * 16 + row;
    *reinterpret_cast<v8bf*>(X + grow * E_ + h * 64 + ch * 8) = pk;
  }
}

// ---------------------------------------------------------------------------
// Kernel 3: out projection + LayerNorm.  Block owns 16 full output rows;
// X-panel staged in LDS (16KB) + 33KB y row buffer; 2x-unrolled ping-pong
// GEMM; block-wide LN reduce + coalesced f32 store.
// ---------------------------------------------------------------------------
#define YPITCH 520
__global__ void k_out_ln(const __bf16* __restrict__ X, const __bf16* __restrict__ Wo,
                         const float* __restrict__ bo, const float* __restrict__ g,
                         const float* __restrict__ bb, float* __restrict__ out) {
  __shared__ __align__(16) __bf16 xp[16 * E_];   // 16KB
  __shared__ float y[16 * YPITCH];               // 33KB
  int lane = threadIdx.x & 31;
  int wave = threadIdx.x >> 5;
  int half = lane >> 4;
  int l15  = lane & 15;
  int mt = blockIdx.x;                       // 0..255

  stage_panel(X + (size_t)mt * 16 * E_, xp);
  const __bf16* xrow = xp + l15 * E_;

  const __bf16* wrow[4];
#pragma unroll
  for (int t = 0; t < 4; ++t) wrow[t] = Wo + (size_t)(wave * 64 + t * 16 + l15) * E_;

  v8f acc[4] = {v8f{}, v8f{}, v8f{}, v8f{}};
  v16bf afA = frag_from_bf16(xrow, 0, half);
  v16bf wfA[4];
#pragma unroll
  for (int t = 0; t < 4; ++t) wfA[t] = frag_from_bf16(wrow[t], 0, half);

  for (int kk = 0; kk < E_; kk += 64) {
    v16bf afB = frag_from_bf16(xrow, kk + 32, half);
    v16bf wfB[4];
#pragma unroll
    for (int t = 0; t < 4; ++t) wfB[t] = frag_from_bf16(wrow[t], kk + 32, half);
#pragma unroll
    for (int t = 0; t < 4; ++t) acc[t] = WMMA_BF16(afA, wfA[t], acc[t]);
    if (kk + 64 < E_) {
      afA = frag_from_bf16(xrow, kk + 64, half);
#pragma unroll
      for (int t = 0; t < 4; ++t) wfA[t] = frag_from_bf16(wrow[t], kk + 64, half);
    }
#pragma unroll
    for (int t = 0; t < 4; ++t) acc[t] = WMMA_BF16(afB, wfB[t], acc[t]);
  }

#pragma unroll
  for (int t = 0; t < 4; ++t) {
    int n = wave * 64 + t * 16 + l15;
    float bias = bo[n];
#pragma unroll
    for (int r = 0; r < 8; ++r)
      y[(r + 8 * half) * YPITCH + n] = acc[t][r] + bias;
  }
  __syncthreads();

  // LayerNorm: 16 threads per row, 32 columns each
  int row = threadIdx.x >> 4;                // 0..15
  int sub = threadIdx.x & 15;
  float s = 0.f, sq = 0.f;
#pragma unroll
  for (int i = 0; i < 32; ++i) {
    float v = y[row * YPITCH + sub * 32 + i];
    s += v; sq += v * v;
  }
#pragma unroll
  for (int m = 1; m < 16; m <<= 1) { s += __shfl_xor(s, m, 32); sq += __shfl_xor(sq, m, 32); }
  float mu  = s * (1.f / 512.f);
  float var = sq * (1.f / 512.f) - mu * mu;
  float rstd = rsqrtf(var + 1e-5f);
  size_t rowg = (size_t)mt * 16 + row;
#pragma unroll
  for (int i = 0; i < 32; ++i) {
    int n = sub * 32 + i;
    out[rowg * E_ + n] = (y[row * YPITCH + n] - mu) * rstd * g[n] + bb[n];
  }
}

// ---------------------------------------------------------------------------
extern "C" void kernel_launch(void* const* d_in, const int* in_sizes, int n_in,
                              void* d_out, int out_size, void* d_ws, size_t ws_size,
                              hipStream_t stream) {
  (void)in_sizes; (void)n_in; (void)out_size; (void)ws_size;
  const float* q   = (const float*)d_in[0];
  const float* k   = (const float*)d_in[1];
  const float* v   = (const float*)d_in[2];
  const float* Wq  = (const float*)d_in[3];
  const float* bq  = (const float*)d_in[4];
  const float* Wk  = (const float*)d_in[5];
  const float* bk  = (const float*)d_in[6];
  const float* Wv  = (const float*)d_in[7];
  const float* bv  = (const float*)d_in[8];
  const float* Wo  = (const float*)d_in[9];
  const float* bo  = (const float*)d_in[10];
  const float* lng = (const float*)d_in[11];
  const float* lnb = (const float*)d_in[12];

  // workspace layout (bf16): activations 3x4MB | weights 4x0.5MB | Qh Kh Vt X
  char* ws = (char*)d_ws;
  const size_t MB = (size_t)1 << 20;
  __bf16* qb  = (__bf16*)(ws);
  __bf16* kb  = (__bf16*)(ws + 4 * MB);
  __bf16* vb  = (__bf16*)(ws + 8 * MB);
  __bf16* Wqb = (__bf16*)(ws + 12 * MB);
  __bf16* Wkb = (__bf16*)(ws + 12 * MB + 512 * 1024);
  __bf16* Wvb = (__bf16*)(ws + 13 * MB);
  __bf16* Wob = (__bf16*)(ws + 13 * MB + 512 * 1024);
  __bf16* Qh  = (__bf16*)(ws + 14 * MB);
  __bf16* Kh  = (__bf16*)(ws + 18 * MB);
  __bf16* Vt  = (__bf16*)(ws + 22 * MB);
  __bf16* X   = (__bf16*)(ws + 26 * MB);

  dim3 blk(256);  // 8 waves/block on wave32

  const int nAct8 = (M_ * E_) / 8;           // 262144
  const int nW8   = (E_ * E_) / 8;           // 32768
  k_cvt<<<dim3((nAct8 + 255) / 256), blk, 0, stream>>>(q, qb, nAct8);
  k_cvt<<<dim3((nAct8 + 255) / 256), blk, 0, stream>>>(k, kb, nAct8);
  k_cvt<<<dim3((nAct8 + 255) / 256), blk, 0, stream>>>(v, vb, nAct8);
  k_cvt<<<dim3((nW8 + 255) / 256),   blk, 0, stream>>>(Wq, Wqb, nW8);
  k_cvt<<<dim3((nW8 + 255) / 256),   blk, 0, stream>>>(Wk, Wkb, nW8);
  k_cvt<<<dim3((nW8 + 255) / 256),   blk, 0, stream>>>(Wv, Wvb, nW8);
  k_cvt<<<dim3((nW8 + 255) / 256),   blk, 0, stream>>>(Wo, Wob, nW8);

  k_proj<<<dim3(256), blk, 0, stream>>>(qb, Wqb, bq, Qh, 0);
  k_proj<<<dim3(256), blk, 0, stream>>>(kb, Wkb, bk, Kh, 0);
  k_proj<<<dim3(256), blk, 0, stream>>>(vb, Wvb, bv, Vt, 1);
  k_attn<<<dim3(256), blk, 0, stream>>>(Qh, Kh, Vt, X);
  k_out_ln<<<dim3(256), blk, 0, stream>>>(X, Wob, bo, lng, lnb, (float*)d_out);
}
